// GQAAttention_48112223650539
// MI455X (gfx1250) — matrix operand order repaired and checked by reading, not compiled, not run on hardware
//
#include <hip/hip_runtime.h>
#include <hip/hip_bf16.h>
#include <math.h>

typedef __attribute__((ext_vector_type(16))) _Float16 v16h;
typedef __attribute__((ext_vector_type(8)))  _Float16 h8;
typedef __attribute__((ext_vector_type(4)))  _Float16 h4;
typedef __attribute__((ext_vector_type(8)))  float    v8f;

union V16 { v16h v; h8 h[2]; };

#define S_   2048
#define E_   4096
#define H_   32
#define KV_  8
#define D_   128
#define NTOT 6144   // E_ + 2*KV_*D_ : fused Q|K|V column space
#define NQ_  4096
#define REP_ 4

__device__ __forceinline__ v8f wmma16x16x32(v16h a, v16h b, v8f c) {
  // D(f32,16x16) = A(f16,16x32) * B(f16,32x16) + C
  return __builtin_amdgcn_wmma_f32_16x16x32_f16(false, a, false, b, (short)0, c,
                                                false, false);
}

// Async DMA: global -> LDS, 16 bytes per lane, tracked by ASYNCcnt.
__device__ __forceinline__ void async_ld_b128(const void* lds_ptr, const void* gptr) {
  const uint32_t lds_addr = (uint32_t)(uintptr_t)lds_ptr;  // aperture: addr[31:0] = LDS offset
  asm volatile("global_load_async_to_lds_b128 %0, %1, off"
               :: "v"(lds_addr), "v"((uint64_t)(uintptr_t)gptr)
               : "memory");
}
__device__ __forceinline__ void wait_async0() {
#if __has_builtin(__builtin_amdgcn_s_wait_asynccnt)
  __builtin_amdgcn_s_wait_asynccnt(0);
#else
  asm volatile("s_wait_asynccnt 0x0" ::: "memory");
#endif
}

// ---------------- k0: convert x (f32) -> xh (f16), same layout ----------------
__global__ void cvt_x_kernel(const float* __restrict__ x, _Float16* __restrict__ xh) {
  const int i = (blockIdx.x * blockDim.x + threadIdx.x) * 4;   // total divides exactly
  float4 v = *(const float4*)(x + i);
  h4 o;
  o[0] = (_Float16)v.x; o[1] = (_Float16)v.y;
  o[2] = (_Float16)v.z; o[3] = (_Float16)v.w;
  *(h4*)(xh + i) = o;
}

// -------- k1: fuse + transpose weights -> wT[n][e] f16 (n in 0..6143) ---------
__global__ void cvt_w_kernel(const float* __restrict__ wq,
                             const float* __restrict__ wk,
                             const float* __restrict__ wv,
                             _Float16* __restrict__ wT) {
  const int idx = blockIdx.x * blockDim.x + threadIdx.x;  // 6144*1024 threads
  const int n  = idx % NTOT;                // fastest -> coalesced source reads
  const int e0 = (idx / NTOT) * 4;
  const float* src; int ncol, nloc;
  if (n < NQ_)            { src = wq; ncol = E_;   nloc = n;        }
  else if (n < NQ_ + 1024){ src = wk; ncol = 1024; nloc = n - NQ_;  }
  else                    { src = wv; ncol = 1024; nloc = n - 5120; }
  _Float16* dst = wT + (size_t)n * E_ + e0;
#pragma unroll
  for (int j = 0; j < 4; ++j)
    dst[j] = (_Float16)src[(size_t)(e0 + j) * ncol + nloc];
}

// ---------------- k2: fused QKV GEMM  Yh = xh @ [wq|wk|wv] + bias -------------
// block = 8 waves over 128 rows x 64 cols; shared B tile staged via async DMA,
// double-buffered in LDS (stride 40 halves: 16B-aligned, bank-spread).
__global__ void gemm_qkv_kernel(const _Float16* __restrict__ xh,
                                const _Float16* __restrict__ wT,
                                const float* __restrict__ bq,
                                const float* __restrict__ bk,
                                const float* __restrict__ bv,
                                _Float16* __restrict__ Yh) {
  __shared__ _Float16 Bs[2][64][40];
  const int tid   = threadIdx.x;
  const int lane  = tid & 31;
  const int wave  = tid >> 5;
  const int l16   = lane & 15;
  const int halfw = lane >> 4;             // 0: lanes 0-15, 1: lanes 16-31
  const int row0  = blockIdx.x * 128 + wave * 16;   // sequence rows
  const int col0  = blockIdx.y * 64;                // fused output columns
  const int brow  = tid >> 2;              // staging: row of B tile (0..63)
  const int bch   = (tid & 3) * 8;         // staging: 8-half chunk within row

  const _Float16* bsrc = wT + (size_t)(col0 + brow) * E_ + bch;

  v8f acc[4] = {{}, {}, {}, {}};
  const _Float16* arow = xh + (size_t)(row0 + l16) * E_;

  // prologue: stage first B tile
  async_ld_b128(&Bs[0][brow][bch], bsrc);
  wait_async0();
  __syncthreads();

  int cur = 0;
  for (int k0 = 0; k0 < E_; k0 += 32) {
    const int next = cur ^ 1;
    if (k0 + 32 < E_)
      async_ld_b128(&Bs[next][brow][bch], bsrc + k0 + 32);

    // A fragment: m = l16, element i -> k = k0 + i + 16*halfw
    V16 a;
    a.h[0] = *(const h8*)(arow + k0 + 16 * halfw);
    a.h[1] = *(const h8*)(arow + k0 + 16 * halfw + 8);
#pragma unroll
    for (int t = 0; t < 4; ++t) {
      // B fragment from LDS: column n = col0+16t+l16, element i -> k = i + 16*halfw
      V16 b;
      b.h[0] = *(const h8*)&Bs[cur][t * 16 + l16][16 * halfw];
      b.h[1] = *(const h8*)&Bs[cur][t * 16 + l16][16 * halfw + 8];
      acc[t] = wmma16x16x32(a.v, b.v, acc[t]);
    }
    wait_async0();
    __syncthreads();
    cur = next;
  }

#pragma unroll
  for (int t = 0; t < 4; ++t) {
    const int n = col0 + t * 16 + l16;
    float bias;
    if (n < NQ_)             bias = bq[n];
    else if (n < NQ_ + 1024) bias = bk[n - NQ_];
    else                     bias = bv[n - 5120];
#pragma unroll
    for (int r = 0; r < 8; ++r) {
      const int m = row0 + r + 8 * halfw;   // C layout: vgpr r -> row r + 8*halfw
      Yh[(size_t)m * NTOT + n] = (_Float16)(acc[t][r] + bias);
    }
  }
}

// ---------------- k3: RoPE in place on Q and K regions of Yh ------------------
__global__ void rope_kernel(_Float16* __restrict__ Yh) {
  const int idx = blockIdx.x * blockDim.x + threadIdx.x;  // 2048 * 2560 threads
  const int s = idx / 2560;
  const int p = idx % 2560;
  int col, d;
  if (p < 2048) { d = p & 63; col = (p >> 6) * D_ + d; }                 // Q heads
  else { const int pk = p - 2048; d = pk & 63; col = NQ_ + (pk >> 6) * D_ + d; } // K heads
  // inv_freq = 10000^(-d/64) ; ln(10000)/64 = 0.14391156831212787
  const float ang = (float)s * __expf(-(float)d * 0.14391156831212787f);
  float sn, cs;
  sincosf(ang, &sn, &cs);
  const size_t base = (size_t)s * NTOT + col;
  const float t1 = (float)Yh[base];
  const float t2 = (float)Yh[base + 64];
  Yh[base]      = (_Float16)(t1 * cs - t2 * sn);
  Yh[base + 64] = (_Float16)(t1 * sn + t2 * cs);
}

// ---------------- k4: transpose V region of Yh -> vT[c][s] --------------------
__global__ void transpose_v_kernel(const _Float16* __restrict__ Yh,
                                   _Float16* __restrict__ vT) {
  const int idx = blockIdx.x * blockDim.x + threadIdx.x;  // 1024*2048 threads
  const int c = idx / S_;
  const int s = idx % S_;                                  // coalesced writes
  vT[(size_t)c * S_ + s] = Yh[(size_t)s * NTOT + 5120 + c];
}

// ---------------- k5: flash attention, one 16-query tile per wave -------------
// K and V chunks are shared by all 8 waves of the block (one head per block),
// so they are staged once per block via async DMA, double-buffered in LDS.
__global__ void attn_kernel(const _Float16* __restrict__ Yh,
                            const _Float16* __restrict__ vT,
                            float* __restrict__ out) {
  __shared__ _Float16 Ks[2][32][136];   // stride 136 halves: conflict-free b128 reads
  __shared__ _Float16 Vs[2][128][40];
  __shared__ _Float16 Pl[8][16 * 48];   // per-wave P staging, stride 48

  const int tid   = threadIdx.x;
  const int lane  = tid & 31;
  const int wave  = tid >> 5;
  const int l16   = lane & 15;
  const int halfw = lane >> 4;
  const int h  = blockIdx.y;
  const int q0 = blockIdx.x * 128 + wave * 16;
  const int kvbase = NQ_ + (h / REP_) * D_;       // K column base in Yh
  const int vbase0 = (h / REP_) * D_;             // V row base in vT

  _Float16* pbuf = &Pl[wave][0];

  // Staging coordinates: 512 16B-chunks of K and of V per 32-key step -> 2 each/thread
  const int krow0 = tid >> 4,        kch0 = (tid & 15) * 8;          // K chunk 0 (rows 0..15)
  const int krow1 = (tid >> 4) + 16, kch1 = kch0;                    // K chunk 1 (rows 16..31)
  const int vrow0 = tid >> 2,        vch0 = (tid & 3) * 8;           // V rows 0..63
  const int vrow1 = (tid >> 2) + 64, vch1 = vch0;                    // V rows 64..127

  const _Float16* ksrc = Yh + kvbase;
  const _Float16* vsrc = vT + (size_t)vbase0 * S_;

  // Preload Q A-fragments for all four 32-wide d-steps
  const _Float16* qrow = Yh + (size_t)(q0 + l16) * NTOT + h * D_;
  V16 qa[4];
#pragma unroll
  for (int d = 0; d < 4; ++d) {
    qa[d].h[0] = *(const h8*)(qrow + d * 32 + 16 * halfw);
    qa[d].h[1] = *(const h8*)(qrow + d * 32 + 16 * halfw + 8);
  }

  v8f O[8] = {{}, {}, {}, {}, {}, {}, {}, {}};
  float mrow[8], lrow[8];
#pragma unroll
  for (int r = 0; r < 8; ++r) { mrow[r] = -INFINITY; lrow[r] = 0.f; }
  const float scale = 0.08838834764831845f;       // 1/sqrt(128)

  // prologue: stage first K/V chunk
  async_ld_b128(&Ks[0][krow0][kch0], ksrc + (size_t)krow0 * NTOT + kch0);
  async_ld_b128(&Ks[0][krow1][kch1], ksrc + (size_t)krow1 * NTOT + kch1);
  async_ld_b128(&Vs[0][vrow0][vch0], vsrc + (size_t)vrow0 * S_ + vch0);
  async_ld_b128(&Vs[0][vrow1][vch1], vsrc + (size_t)vrow1 * S_ + vch1);
  wait_async0();
  __syncthreads();

  int cur = 0;
  for (int kb = 0; kb < S_; kb += 32) {
    const int next = cur ^ 1;
    if (kb + 32 < S_) {
      const int kn = kb + 32;
      async_ld_b128(&Ks[next][krow0][kch0], ksrc + (size_t)(kn + krow0) * NTOT + kch0);
      async_ld_b128(&Ks[next][krow1][kch1], ksrc + (size_t)(kn + krow1) * NTOT + kch1);
      async_ld_b128(&Vs[next][vrow0][vch0], vsrc + (size_t)vrow0 * S_ + kn + vch0);
      async_ld_b128(&Vs[next][vrow1][vch1], vsrc + (size_t)vrow1 * S_ + kn + vch1);
    }

    // ---- scores: two 16x16 tiles (keys kb..kb+15, kb+16..kb+31), K from LDS ----
    v8f s0 = {}, s1 = {};
#pragma unroll
    for (int d = 0; d < 4; ++d) {
      V16 b0, b1;
      b0.h[0] = *(const h8*)&Ks[cur][l16][d * 32 + 16 * halfw];
      b0.h[1] = *(const h8*)&Ks[cur][l16][d * 32 + 16 * halfw + 8];
      s0 = wmma16x16x32(qa[d].v, b0.v, s0);
      b1.h[0] = *(const h8*)&Ks[cur][16 + l16][d * 32 + 16 * halfw];
      b1.h[1] = *(const h8*)&Ks[cur][16 + l16][d * 32 + 16 * halfw + 8];
      s1 = wmma16x16x32(qa[d].v, b1.v, s1);
    }
    // ---- online softmax (rows live per lane-half; reduce over 16 lanes) ----
#pragma unroll
    for (int r = 0; r < 8; ++r) {
      s0[r] *= scale; s1[r] *= scale;
      float mx = fmaxf(s0[r], s1[r]);
      for (int off = 8; off; off >>= 1) mx = fmaxf(mx, __shfl_xor(mx, off, 32));
      const float mnew = fmaxf(mrow[r], mx);
      const float corr = __expf(mrow[r] - mnew);
      mrow[r] = mnew;
      const float p0 = __expf(s0[r] - mnew);
      const float p1 = __expf(s1[r] - mnew);
      s0[r] = p0; s1[r] = p1;
      float rs = p0 + p1;
      for (int off = 8; off; off >>= 1) rs += __shfl_xor(rs, off, 32);
      lrow[r] = lrow[r] * corr + rs;
#pragma unroll
      for (int t = 0; t < 8; ++t) O[t][r] *= corr;
    }
    // ---- stage P (C layout) into wave-private LDS, reload as A fragment ----
#pragma unroll
    for (int r = 0; r < 8; ++r) {
      const int m = r + 8 * halfw;
      pbuf[m * 48 + l16]      = (_Float16)s0[r];
      pbuf[m * 48 + 16 + l16] = (_Float16)s1[r];
    }
    V16 pa;
    pa.h[0] = *(const h8*)(pbuf + l16 * 48 + 16 * halfw);
    pa.h[1] = *(const h8*)(pbuf + l16 * 48 + 16 * halfw + 8);
    // ---- O += P @ V over all 128 head dims (8 column tiles), V from LDS ----
#pragma unroll
    for (int t = 0; t < 8; ++t) {
      V16 vb;
      vb.h[0] = *(const h8*)&Vs[cur][t * 16 + l16][16 * halfw];
      vb.h[1] = *(const h8*)&Vs[cur][t * 16 + l16][16 * halfw + 8];
      O[t] = wmma16x16x32(pa.v, vb.v, O[t]);
    }

    wait_async0();
    __syncthreads();
    cur = next;
  }

  // ---- normalize and write out[s][h*128 + d] (f32) ----
#pragma unroll
  for (int t = 0; t < 8; ++t) {
    const int e = h * D_ + t * 16 + l16;
#pragma unroll
    for (int r = 0; r < 8; ++r) {
      const int m = q0 + r + 8 * halfw;
      out[(size_t)m * E_ + e] = O[t][r] / lrow[r];
    }
  }
}

extern "C" void kernel_launch(void* const* d_in, const int* in_sizes, int n_in,
                              void* d_out, int out_size, void* d_ws, size_t ws_size,
                              hipStream_t stream) {
  (void)in_sizes; (void)n_in; (void)out_size; (void)ws_size;
  const float* x  = (const float*)d_in[0];
  const float* wq = (const float*)d_in[1];
  const float* bq = (const float*)d_in[2];
  const float* wk = (const float*)d_in[3];
  const float* bk = (const float*)d_in[4];
  const float* wv = (const float*)d_in[5];
  const float* bv = (const float*)d_in[6];
  float* out = (float*)d_out;

  _Float16* xh = (_Float16*)d_ws;                       // 2048*4096     (16 MB)
  _Float16* wT = xh + (size_t)S_ * E_;                  // 6144*4096     (48 MB)
  _Float16* Yh = wT + (size_t)NTOT * E_;                // 2048*6144     (24 MB)
  _Float16* vT = Yh + (size_t)S_ * NTOT;                // 1024*2048     ( 4 MB)

  cvt_x_kernel      <<<dim3((S_ * E_) / (256 * 4)), 256, 0, stream>>>(x, xh);
  cvt_w_kernel      <<<dim3((NTOT * E_) / (256 * 4)), 256, 0, stream>>>(wq, wk, wv, wT);
  gemm_qkv_kernel   <<<dim3(S_ / 128, NTOT / 64), 256, 0, stream>>>(xh, wT, bq, bk, bv, Yh);
  rope_kernel       <<<dim3((S_ * 2560) / 256), 256, 0, stream>>>(Yh);
  transpose_v_kernel<<<dim3((1024 * S_) / 256), 256, 0, stream>>>(Yh, vT);
  attn_kernel       <<<dim3(S_ / 128, H_), 256, 0, stream>>>(Yh, vT, out);
}
